// MaskAttention_65893388255524
// MI455X (gfx1250) — matrix operand order repaired and checked by reading, not compiled, not run on hardware
//
#include <hip/hip_runtime.h>
#include <hip/hip_bf16.h>

// MI455X (gfx1250) flash-attention implementation.
// B=4, C=128, H=W=64, N=4096. GEMMs via v_wmma_f32_16x16x32_f16;
// K/V tile staging via Tensor Data Mover (tensor_load_to_lds) when available.

#define BB 4
#define CC 128
#define NN 4096
#define TQ 64   // query tile per workgroup (4 waves x 16 queries)
#define TK 64   // key tile per iteration

typedef __attribute__((ext_vector_type(16))) _Float16 v16h;
typedef __attribute__((ext_vector_type(8)))  _Float16 v8h;
typedef __attribute__((ext_vector_type(8)))  float    v8f;
typedef __attribute__((ext_vector_type(4)))  unsigned int u32x4;
typedef __attribute__((ext_vector_type(8)))  int      i32x8;
typedef __attribute__((ext_vector_type(4)))  int      i32x4;

#if defined(__has_builtin)
#if __has_builtin(__builtin_amdgcn_tensor_load_to_lds) && \
    __has_builtin(__builtin_amdgcn_s_wait_tensorcnt)
#define HAVE_TDM 1
#endif
#endif
#ifndef HAVE_TDM
#define HAVE_TDM 0
#endif

// Build a 16-half WMMA fragment from two 16-byte chunks (LDS or global).
__device__ __forceinline__ v16h frag16(const _Float16* p0, const _Float16* p1) {
    v8h lo = *(const v8h*)p0;
    v8h hh = *(const v8h*)p1;
    v16h r;
#pragma unroll
    for (int i = 0; i < 8; ++i) { r[i] = lo[i]; r[8 + i] = hh[i]; }
    return r;
}

#if HAVE_TDM
// Issue one 2-D TENSOR_LOAD_TO_LDS: tile_d1 rows of tile_d0 16-bit elements,
// global rows strided by stride0 elements, packed contiguously into LDS.
// Descriptor layout per CDNA5 ISA.
__device__ __forceinline__ void tdm_tile_2d(unsigned int lds_off,
                                            const _Float16* gp,
                                            unsigned int tile_d0,
                                            unsigned int tile_d1,
                                            unsigned long long stride0) {
    const unsigned long long ga = (unsigned long long)(uintptr_t)gp;
    u32x4 g0;
    g0.x = 1u;                                   // count=1, user mode, no gather
    g0.y = lds_off;                              // lds_addr (bytes)
    g0.z = (unsigned int)ga;                     // global_addr[31:0]
    g0.w = (unsigned int)((ga >> 32) & 0x01FFFFFFu) | (2u << 30); // addr[56:32] | type=2
    const unsigned int td0 = 1u << 30, td1 = 1u << 30;  // huge tensor dims: no OOB clip
    i32x8 g1;
    g1[0] = 0x00010000;                          // wg_mask=0, data_size=1 (2 bytes)
    g1[1] = (int)((td0 & 0xFFFFu) << 16);        // tensor_dim0[15:0]
    g1[2] = (int)((td0 >> 16) | ((td1 & 0xFFFFu) << 16));
    g1[3] = (int)((td1 >> 16) | (tile_d0 << 16)); // tile_dim0
    g1[4] = (int)tile_d1;                        // tile_dim1; tile_dim2=0
    g1[5] = (int)(unsigned int)stride0;          // tensor_dim0_stride[31:0]
    g1[6] = (int)(unsigned int)((stride0 >> 32) & 0xFFFFu);
    g1[7] = 0;
    i32x4 z4 = {};
#if __clang_major__ >= 23
    i32x8 z8 = {};
    __builtin_amdgcn_tensor_load_to_lds(g0, g1, z4, z4, z8, 0);
#else
    __builtin_amdgcn_tensor_load_to_lds(g0, g1, z4, z4, 0);
#endif
}
#endif

// ---------------------------------------------------------------------------
// Kernel 1: QKV projections (p-loop fully unrolled -> branch-free stores).
//   Q = prompt_tokens @ Wq^T + bq   -> token-major f16 (B,N,C)
//   K = x_tokens      @ Wk^T + bk   -> token-major f16 (B,N,C)
//   V = x_tokens      @ Wv^T + bv   -> chan-major  f16 (B,C,N)
// ---------------------------------------------------------------------------
__global__ __launch_bounds__(128) void qkv_proj_kernel(
    const float* __restrict__ prompt, const float* __restrict__ x,
    const float* __restrict__ Wq, const float* __restrict__ bq,
    const float* __restrict__ Wk, const float* __restrict__ bk,
    const float* __restrict__ Wv, const float* __restrict__ bv,
    _Float16* __restrict__ Qh, _Float16* __restrict__ Kh,
    _Float16* __restrict__ Vh)
{
    __shared__ _Float16 sX[TQ * CC];   // x tile, [token][chan]
    __shared__ _Float16 sP[TQ * CC];   // prompt tile, [token][chan]
    __shared__ _Float16 sW[CC * CC];   // current weight, [cout][cin]

    const int tid  = threadIdx.x;
    const int b    = blockIdx.x >> 6;
    const int t0   = (blockIdx.x & 63) * TQ;
    const int lane = tid & 31;
    const int w    = tid >> 5;       // wave 0..3 -> 16 tokens each
    const int q    = lane & 15;
    const int hi   = lane >> 4;

    // Stage input tiles (transpose (C,N)->(token,chan)); coalesced along n.
    {
        const int tok = tid & 63;
        const int c0  = tid >> 6;     // 0 or 1
#pragma unroll 4
        for (int k = 0; k < 64; ++k) {
            const int c = k * 2 + c0;
            sX[tok * CC + c] = (_Float16)x[(b * CC + c) * NN + t0 + tok];
            sP[tok * CC + c] = (_Float16)prompt[(b * CC + c) * NN + t0 + tok];
        }
    }

    const float* Wsrc[3] = { Wq, Wk, Wv };
    const float* Bsrc[3] = { bq, bk, bv };

#pragma unroll
    for (int p = 0; p < 3; ++p) {
        __syncthreads();
        for (int i = tid; i < CC * CC; i += 128)
            sW[i] = (_Float16)Wsrc[p][i];
        __syncthreads();

        const _Float16* src = (p == 0) ? sP : sX;

#pragma unroll
        for (int cg = 0; cg < 8; ++cg) {
            const int cout = cg * 16 + q;
            const float bias = Bsrc[p][cout];
            v8f acc;
#pragma unroll
            for (int r = 0; r < 8; ++r) acc[r] = bias;

#pragma unroll
            for (int kk = 0; kk < 4; ++kk) {
                // A: 16 tokens x 32 chans (rows = lane&15)
                const _Float16* ab = src + (w * 16 + q) * CC + kk * 32 + hi * 16;
                v16h a = frag16(ab, ab + 8);
                // B: 32 cin x 16 cout; contiguous 16 halfs per lane from row-major W
                const _Float16* bbp = sW + cout * CC + kk * 32 + hi * 16;
                v16h bf = frag16(bbp, bbp + 8);
                acc = __builtin_amdgcn_wmma_f32_16x16x32_f16(
                        false, a, false, bf, (short)0, acc, false, false);
            }

#pragma unroll
            for (int r = 0; r < 8; ++r) {
                const int tok = t0 + w * 16 + r + hi * 8;
                const _Float16 h = (_Float16)acc[r];
                if (p == 0)      Qh[(b * NN + tok) * CC + cout] = h;  // folds at
                else if (p == 1) Kh[(b * NN + tok) * CC + cout] = h;  // compile
                else             Vh[(b * CC + cout) * NN + tok] = h;  // time
            }
        }
    }
}

// ---------------------------------------------------------------------------
// Kernel 2: flash attention (S^T = K Q^T, O^T = V^T P^T), masked online
// softmax, residual, LayerNorm. K/V tiles double-buffered in LDS, staged by
// the Tensor Data Mover (wave 0 issues, s_wait_tensorcnt pipelines kt+1's
// DMA behind kt's WMMA work).
// ---------------------------------------------------------------------------
__global__ __launch_bounds__(128) void flash_attn_ln_kernel(
    const _Float16* __restrict__ Qh, const _Float16* __restrict__ Kh,
    const _Float16* __restrict__ Vh, const float* __restrict__ x,
    const float* __restrict__ gamma, const float* __restrict__ beta,
    const int* __restrict__ mask_bits, float* __restrict__ out)
{
    __shared__ _Float16 sK[2][TK * CC];   // K tile  [key][chan]   2 x 16KB
    __shared__ _Float16 sV[2][CC * TK];   // V tile  [chan][key]   2 x 16KB
    __shared__ float    sMask[TK];
    __shared__ float    sG[CC], sBt[CC];

    const int tid  = threadIdx.x;
    const int b    = blockIdx.x >> 6;
    const int q0   = (blockIdx.x & 63) * TQ;
    const int lane = tid & 31;
    const int w    = tid >> 5;
    const int q    = lane & 15;        // this lane's query column
    const int hi   = lane >> 4;

    if (tid < CC) { sG[tid] = gamma[tid]; sBt[tid] = beta[tid]; }

    // Q^T B-fragments for this wave's 16 queries: kept in registers.
    v16h Qb[4];
    {
        const _Float16* qb = Qh + ((size_t)b * NN + q0 + w * 16 + q) * CC;
#pragma unroll
        for (int kk = 0; kk < 4; ++kk) {
            const _Float16* pz = qb + kk * 32 + hi * 16;
            Qb[kk] = frag16(pz, pz + 8);
        }
    }

    float m_run = -1e30f, l_run = 0.0f;
    v8f O[8];
#pragma unroll
    for (int cg = 0; cg < 8; ++cg)
#pragma unroll
        for (int r = 0; r < 8; ++r) O[cg][r] = 0.0f;

    const float rsC = 0.08838834764831845f;   // 1/sqrt(128)
    const int nt = NN / TK;

#if HAVE_TDM
    if (w == 0) {
        // Preload tile 0 into buffer 0 (K: 64 rows x 256B; V: 128 rows x 128B).
        tdm_tile_2d((unsigned int)(uintptr_t)&sK[0][0],
                    Kh + (size_t)b * NN * CC, CC, TK, CC);
        tdm_tile_2d((unsigned int)(uintptr_t)&sV[0][0],
                    Vh + (size_t)b * CC * NN, TK, CC, NN);
    }
#endif

    for (int kt = 0; kt < nt; ++kt) {
        const int k0  = kt * TK;
        const int buf = kt & 1;
        __syncthreads();   // all waves done reading buf (from iteration kt-2)

        if (tid < TK)
            sMask[tid] = (mask_bits[b * NN + k0 + tid] > 0) ? 0.0f : -1e30f;

#if HAVE_TDM
        if (w == 0) {
            if (kt + 1 < nt) {
                const int nb = (kt + 1) & 1;
                const int nk = k0 + TK;
                tdm_tile_2d((unsigned int)(uintptr_t)&sK[nb][0],
                            Kh + ((size_t)b * NN + nk) * CC, CC, TK, CC);
                tdm_tile_2d((unsigned int)(uintptr_t)&sV[nb][0],
                            Vh + (size_t)b * CC * NN + nk, TK, CC, NN);
                __builtin_amdgcn_s_wait_tensorcnt(2);  // tile kt's pair complete
            } else {
                __builtin_amdgcn_s_wait_tensorcnt(0);
            }
        }
#else
        {
            const v8h* ks = (const v8h*)(Kh + ((size_t)b * NN + k0) * CC);
            v8h* kd = (v8h*)sK[buf];
            for (int i = tid; i < TK * CC / 8; i += 128) kd[i] = ks[i];
            v8h* vd = (v8h*)sV[buf];
            for (int i = tid; i < CC * TK / 8; i += 128) {
                const int c = i >> 3, seg = i & 7;
                vd[i] = *(const v8h*)(Vh + ((size_t)b * CC + c) * NN + k0 + seg * 8);
            }
        }
#endif
        __syncthreads();   // tile kt resident in LDS for all waves

        const _Float16* sKc = sK[buf];
        const _Float16* sVc = sV[buf];

        // S^T[key,query] for 4 key groups of 16.
        v8f S[4];
#pragma unroll
        for (int g = 0; g < 4; ++g) {
            v8f acc;
#pragma unroll
            for (int r = 0; r < 8; ++r) acc[r] = 0.0f;
#pragma unroll
            for (int kk = 0; kk < 4; ++kk) {
                const _Float16* ab = sKc + (g * 16 + q) * CC + kk * 32 + hi * 16;
                v16h a = frag16(ab, ab + 8);
                acc = __builtin_amdgcn_wmma_f32_16x16x32_f16(
                        false, a, false, Qb[kk], (short)0, acc, false, false);
            }
            S[g] = acc;
        }

        // Scale + additive key mask + per-query tile max (in-lane + 1 shuffle).
        float tmax = -1e30f;
#pragma unroll
        for (int g = 0; g < 4; ++g)
#pragma unroll
            for (int r = 0; r < 8; ++r) {
                float s = S[g][r] * rsC + sMask[g * 16 + r + hi * 8];
                S[g][r] = s;
                tmax = fmaxf(tmax, s);
            }
        tmax = fmaxf(tmax, __shfl_xor(tmax, 16));

        const float m_new = fmaxf(m_run, tmax);
        const float alpha = __expf(m_run - m_new);
        float rsum = 0.0f;
#pragma unroll
        for (int g = 0; g < 4; ++g)
#pragma unroll
            for (int r = 0; r < 8; ++r) {
                const float p = __expf(S[g][r] - m_new);
                S[g][r] = p;
                rsum += p;
            }
        rsum += __shfl_xor(rsum, 16);
        l_run = l_run * alpha + rsum;
        m_run = m_new;
#pragma unroll
        for (int cg = 0; cg < 8; ++cg)
#pragma unroll
            for (int r = 0; r < 8; ++r) O[cg][r] *= alpha;

        // O^T += V^T * P^T. P^T B-fragments built in registers via half-wave
        // shuffles (D-frag layout -> B-frag layout transpose).
#pragma unroll
        for (int kc = 0; kc < 2; ++kc) {
            v16h Bp;
#pragma unroll
            for (int j = 0; j < 8; ++j) {
                const float d0 = S[2 * kc][j];
                const float d1 = S[2 * kc + 1][j];
                const float r1 = __shfl_xor(d1, 16);
                const float r0 = __shfl_xor(d0, 16);
                Bp[j]     = (_Float16)(hi ? r1 : d0);
                Bp[8 + j] = (_Float16)(hi ? d1 : r0);
            }
#pragma unroll
            for (int cg = 0; cg < 8; ++cg) {
                const _Float16* ab = sVc + (cg * 16 + q) * TK + kc * 32 + hi * 16;
                v16h a = frag16(ab, ab + 8);
                O[cg] = __builtin_amdgcn_wmma_f32_16x16x32_f16(
                        false, a, false, Bp, (short)0, O[cg], false, false);
            }
        }
    }

    // Epilogue: normalize by softmax denom, residual add, LayerNorm, store.
    const float inv_l = 1.0f / l_run;
    const int tok = q0 + w * 16 + q;
    float s1 = 0.0f, s2 = 0.0f;
#pragma unroll
    for (int cg = 0; cg < 8; ++cg)
#pragma unroll
        for (int r = 0; r < 8; ++r) {
            const int ch = cg * 16 + r + hi * 8;
            const float v = O[cg][r] * inv_l + x[((size_t)b * CC + ch) * NN + tok];
            O[cg][r] = v;
            s1 += v;
            s2 += v * v;
        }
    s1 += __shfl_xor(s1, 16);
    s2 += __shfl_xor(s2, 16);
    const float mu   = s1 * (1.0f / 128.0f);
    const float var  = s2 * (1.0f / 128.0f) - mu * mu;
    const float rstd = rsqrtf(var + 1e-5f);
#pragma unroll
    for (int cg = 0; cg < 8; ++cg)
#pragma unroll
        for (int r = 0; r < 8; ++r) {
            const int ch = cg * 16 + r + hi * 8;
            out[((size_t)b * CC + ch) * NN + tok] =
                (O[cg][r] - mu) * rstd * sG[ch] + sBt[ch];
        }
}

// ---------------------------------------------------------------------------
extern "C" void kernel_launch(void* const* d_in, const int* in_sizes, int n_in,
                              void* d_out, int out_size, void* d_ws, size_t ws_size,
                              hipStream_t stream) {
    (void)in_sizes; (void)n_in; (void)out_size; (void)ws_size;

    const float* prompt = (const float*)d_in[0];
    const float* x      = (const float*)d_in[1];
    const float* Wq     = (const float*)d_in[2];
    const float* bq     = (const float*)d_in[3];
    const float* Wk     = (const float*)d_in[4];
    const float* bk     = (const float*)d_in[5];
    const float* Wv     = (const float*)d_in[6];
    const float* bv     = (const float*)d_in[7];
    const float* gamma  = (const float*)d_in[8];
    const float* beta   = (const float*)d_in[9];
    const int*   mask   = (const int*)d_in[10];
    float* out = (float*)d_out;

    _Float16* Qh = (_Float16*)d_ws;                 // 4 MB
    _Float16* Kh = Qh + (size_t)BB * NN * CC;       // 4 MB
    _Float16* Vh = Kh + (size_t)BB * NN * CC;       // 4 MB

    dim3 grid(BB * (NN / TQ));   // 256 workgroups
    dim3 block(128);             // 4 waves (wave32)

    qkv_proj_kernel<<<grid, block, 0, stream>>>(
        prompt, x, Wq, bq, Wk, bk, Wv, bv, Qh, Kh, Vh);
    flash_attn_ln_kernel<<<grid, block, 0, stream>>>(
        Qh, Kh, Vh, x, gamma, beta, mask, out);
}
